// ContrastiveLoss_44513041056397
// MI455X (gfx1250) — matrix operand order repaired and checked by reading, not compiled, not run on hardware
//
#include <hip/hip_runtime.h>
#include <hip/hip_bf16.h>
#include <math.h>

// ---------------------------------------------------------------------------
// Contrastive loss, fused:  z = normalize(emb);  S = z[:n] @ z.T / 0.5
// loss = sum_i (n-1-i)*log(sum_j exp(S_ij) - exp(S_ii)) - sum_{i<j<n} S_ij
// out  = -2/n*(n-1)*loss
//
// bf16 hi/lo split (bf16x3) on v_wmma_f32_16x16x32_bf16.
// - 8 column tiles per wave: one LDS A-fragment feeds 24 WMMAs per K-step.
// - Wave-uniform tile bases (readfirstlane) -> SGPR-based global loads with a
//   single shared per-lane offset VGPR (kills the 64-bit pointer-add storm).
// - A-strip staged into LDS via CDNA5 global_load_async_to_lds_b128.
// ---------------------------------------------------------------------------

typedef __attribute__((ext_vector_type(16))) __bf16 bf16x16;
typedef __attribute__((ext_vector_type(8)))  __bf16 bf16x8;
typedef __attribute__((ext_vector_type(8)))  float  f32x8;

#define DIMK   1024
#define NTOT   8192
#define NROWS  2048              // n = B/4
#define TEMP_INV 2.0f            // 1 / TEMPERATURE
#define LDA    (DIMK + 8)        // padded LDS row stride (bf16 elems)
#define NT     8                 // column tiles per wave per group
#define NGROUP 4                 // 32 tiles per wave total

// ---------------------------------------------------------------- normalize
__global__ void cl_normalize_split(const float* __restrict__ x,
                                   __bf16* __restrict__ zhi,
                                   __bf16* __restrict__ zlo) {
  __shared__ float wsum[8];
  __shared__ float rnorm_s;
  const int row = blockIdx.x;
  const int t   = threadIdx.x;                 // 256 threads
  const float* xr = x + (size_t)row * DIMK;

  float v[4];
  float s = 0.f;
#pragma unroll
  for (int i = 0; i < 4; ++i) { v[i] = xr[t + 256 * i]; s += v[i] * v[i]; }
#pragma unroll
  for (int off = 16; off >= 1; off >>= 1) s += __shfl_xor(s, off, 32);
  if ((t & 31) == 0) wsum[t >> 5] = s;
  __syncthreads();
  if (t == 0) {
    float tot = 0.f;
    for (int w = 0; w < 8; ++w) tot += wsum[w];
    rnorm_s = 1.0f / sqrtf(tot);
  }
  __syncthreads();
  const float rn = rnorm_s;
#pragma unroll
  for (int i = 0; i < 4; ++i) {
    float z  = v[i] * rn;
    __bf16 h = (__bf16)z;
    __bf16 l = (__bf16)(z - (float)h);
    zhi[(size_t)row * DIMK + t + 256 * i] = h;
    zlo[(size_t)row * DIMK + t + 256 * i] = l;
  }
}

// ------------------------------------------------------------ fused GEMM
// grid: 128 blocks (16-row strips of the first 2048 rows), 512 thr = 16 waves.
__global__ void __launch_bounds__(512, 1)
cl_gemm_fused(const __bf16* __restrict__ zhi,
              const __bf16* __restrict__ zlo,
              float* __restrict__ acc_out) {
  __shared__ __bf16 Ahi[16 * LDA];
  __shared__ __bf16 Alo[16 * LDA];
  __shared__ float rowExp[16];
  __shared__ float rowDiag[16];
  __shared__ float simShared;

  const int tid  = threadIdx.x;
  const int wave = __builtin_amdgcn_readfirstlane(tid >> 5);  // wave-uniform -> SGPR
  const int lane = tid & 31;
  const int i0   = blockIdx.x * 16;

  if (tid < 16) { rowExp[tid] = 0.f; rowDiag[tid] = 0.f; }
  if (tid == 0) simShared = 0.f;

  // Stage the 16x1024 A strip (hi+lo) into LDS via CDNA5 async DMA path.
  // Low 32 bits of a flat shared-pointer == LDS byte address (ISA 10.2).
  for (int e = tid; e < 16 * (DIMK / 8); e += 512) {
    const int r = e / (DIMK / 8);
    const int c = (e % (DIMK / 8)) * 8;
    unsigned lds_hi = (unsigned)(size_t)(&Ahi[r * LDA + c]);
    unsigned lds_lo = (unsigned)(size_t)(&Alo[r * LDA + c]);
    const __bf16* ghi = zhi + (size_t)(i0 + r) * DIMK + c;
    const __bf16* glo = zlo + (size_t)(i0 + r) * DIMK + c;
    asm volatile("global_load_async_to_lds_b128 %0, %1, off"
                 :: "v"(lds_hi), "v"(ghi) : "memory");
    asm volatile("global_load_async_to_lds_b128 %0, %1, off"
                 :: "v"(lds_lo), "v"(glo) : "memory");
  }
  asm volatile("s_wait_asynccnt 0x0" ::: "memory");
  __syncthreads();

  const int l16   = lane & 15;
  const int khalf = lane >> 4;
  const int aoff  = l16 * LDA + khalf * 16;     // A LDS per-lane base (elems)
  const int voff  = l16 * DIMK + khalf * 16;    // B per-lane offset (elems), shared by all tiles

  float simAcc = 0.f;
  float eAcc[8], dAcc[8];
#pragma unroll
  for (int v = 0; v < 8; ++v) { eAcc[v] = 0.f; dAcc[v] = 0.f; }

#pragma unroll 1
  for (int g = 0; g < NGROUP; ++g) {
    f32x8 acc[NT];
    int   jb[NT];
    const __bf16* bbh[NT];   // wave-uniform tile bases -> SGPR pairs
    const __bf16* bbl[NT];
#pragma unroll
    for (int nt = 0; nt < NT; ++nt) {
      acc[nt] = (f32x8){0.f, 0.f, 0.f, 0.f, 0.f, 0.f, 0.f, 0.f};
      jb[nt]  = ((g * NT + nt) * 16 + wave) * 16;          // scalar tile column base
      bbh[nt] = zhi + (size_t)jb[nt] * DIMK;
      bbl[nt] = zlo + (size_t)jb[nt] * DIMK;
    }

#pragma unroll 2
    for (int k0 = 0; k0 < DIMK; k0 += 32) {
      // A fragment from LDS: lane = row M, K = k0 + khalf*16 + [0..15]
      union { bf16x16 v; bf16x8 h[2]; } afh, afl;
      afh.h[0] = *(const bf16x8*)(&Ahi[aoff + k0]);
      afh.h[1] = *(const bf16x8*)(&Ahi[aoff + k0 + 8]);
      afl.h[0] = *(const bf16x8*)(&Alo[aoff + k0]);
      afl.h[1] = *(const bf16x8*)(&Alo[aoff + k0 + 8]);

#pragma unroll
      for (int nt = 0; nt < NT; ++nt) {
        // B fragment: lane = column N, K = k0 + khalf*16 + [0..15]
        union { bf16x16 v; bf16x8 h[2]; } bfh, bfl;
        bfh.h[0] = *(const bf16x8*)(bbh[nt] + voff + k0);
        bfh.h[1] = *(const bf16x8*)(bbh[nt] + voff + k0 + 8);
        bfl.h[0] = *(const bf16x8*)(bbl[nt] + voff + k0);
        bfl.h[1] = *(const bf16x8*)(bbl[nt] + voff + k0 + 8);

        // bf16x3: hi*hi + hi*lo + lo*hi, f32 accumulate
        acc[nt] = __builtin_amdgcn_wmma_f32_16x16x32_bf16(false, afh.v, false, bfh.v,
                                                          (short)0, acc[nt], false, false);
        acc[nt] = __builtin_amdgcn_wmma_f32_16x16x32_bf16(false, afh.v, false, bfl.v,
                                                          (short)0, acc[nt], false, false);
        acc[nt] = __builtin_amdgcn_wmma_f32_16x16x32_bf16(false, afl.v, false, bfh.v,
                                                          (short)0, acc[nt], false, false);
      }
    }

    // Epilogue: C layout: row = v + 8*khalf, col = lane%16
#pragma unroll
    for (int nt = 0; nt < NT; ++nt) {
      const int j = jb[nt] + l16;
#pragma unroll
      for (int v = 0; v < 8; ++v) {
        const int i   = i0 + v + khalf * 8;
        const float s = acc[nt][v] * TEMP_INV;
        const float e = __expf(s);
        eAcc[v] += e;
        if (j == i) dAcc[v] += e;
        if (j < NROWS && j > i) simAcc += s;
      }
    }
  }

  // Row-exp reduction: lanes within a 16-lane half share the same row.
#pragma unroll
  for (int v = 0; v < 8; ++v) {
    float s = eAcc[v], d = dAcc[v];
#pragma unroll
    for (int off = 8; off >= 1; off >>= 1) {
      s += __shfl_xor(s, off, 32);
      d += __shfl_xor(d, off, 32);
    }
    if (l16 == 0) {
      atomicAdd(&rowExp[v + khalf * 8], s);
      atomicAdd(&rowDiag[v + khalf * 8], d);
    }
  }
#pragma unroll
  for (int off = 16; off >= 1; off >>= 1) simAcc += __shfl_xor(simAcc, off, 32);
  if (lane == 0) atomicAdd(&simShared, simAcc);
  __syncthreads();

  if (tid == 0) {
    float part = -simShared;
    for (int r = 0; r < 16; ++r) {
      const int i = i0 + r;
      const float denom = rowExp[r] - rowDiag[r];
      part += (float)(NROWS - 1 - i) * __logf(denom);
    }
    atomicAdd(acc_out, part);
  }
}

// ------------------------------------------------------------- init / final
__global__ void cl_init(float* acc) { if (threadIdx.x == 0) acc[0] = 0.f; }

__global__ void cl_finalize(const float* __restrict__ acc, float* __restrict__ out) {
  if (threadIdx.x == 0) {
    const float n = (float)NROWS;
    out[0] = (-2.0f / n) * (n - 1.0f) * acc[0];
  }
}

// ---------------------------------------------------------------- launcher
extern "C" void kernel_launch(void* const* d_in, const int* in_sizes, int n_in,
                              void* d_out, int out_size, void* d_ws, size_t ws_size,
                              hipStream_t stream) {
  (void)in_sizes; (void)n_in; (void)out_size; (void)ws_size;
  const float* x = (const float*)d_in[0];
  float* out = (float*)d_out;

  char* ws = (char*)d_ws;
  __bf16* zhi = (__bf16*)ws;                                        // 16 MB
  __bf16* zlo = (__bf16*)(ws + (size_t)NTOT * DIMK * 2);            // 16 MB
  float*  acc = (float*)(ws + (size_t)NTOT * DIMK * 4);             // 4 B

  cl_init<<<1, 1, 0, stream>>>(acc);
  cl_normalize_split<<<NTOT, 256, 0, stream>>>(x, zhi, zlo);
  cl_gemm_fused<<<NROWS / 16, 512, 0, stream>>>(zhi, zlo, acc);
  cl_finalize<<<1, 1, 0, stream>>>(acc, out);
}
